// LinearAttention_68478958568020
// MI455X (gfx1250) — compile-verified, matrix operand order checked by reading, not run on hardware
//
#include <hip/hip_runtime.h>
#include <hip/hip_bf16.h>

// ---------------------------------------------------------------------------
// Causal linear attention (chunked form) for MI455X / gfx1250, wave32 + WMMA.
//   - All GEMMs: f16 inputs, f32 accumulation via v_wmma_f32_16x16x32_f16.
//   - Q/K/V projections fused into one N=3072 GEMM (phi on the first 2048 cols).
//   - Attention recurrence in chunk-parallel form (chunk = 64 = d_k).
// ---------------------------------------------------------------------------

#define D_MODEL 1024
#define QKV_N   3072                    // fused q|k|v output width
#define N_HEADS 16
#define BB      2
#define TT      2048
#define M_TOT   (BB * TT)               // 4096 rows
#define NCHUNK  (TT / 64)               // 32 chunks per (b,h)
#define NTASK   (BB * N_HEADS * NCHUNK) // 1024 chunk tasks

typedef __attribute__((ext_vector_type(16))) _Float16 v16h;
typedef __attribute__((ext_vector_type(8)))  _Float16 h8v;
typedef __attribute__((ext_vector_type(8)))  float    v8f;

// Per-lane base pointer for a 16x32 f16 fragment (CDNA5 wave32 layout):
//   lane L: row = L&15 ; halves 0..7 = K kofs..+7 ; halves 8..15 = K kofs+16..+23
//   kofs = (L<16) ? 0 : 8
__device__ __forceinline__ const _Float16* frag_ptr(const _Float16* p0, int ld) {
  const int lane = threadIdx.x & 31;
  return p0 + (lane & 15) * ld + ((lane >> 4) << 3);
}
__device__ __forceinline__ v16h load_frag_p(const _Float16* p) {
  union { v16h v; struct { h8v lo, hi; } s; } u;
  u.s.lo = *(const h8v*)(p);
  u.s.hi = *(const h8v*)(p + 16);
  return u.v;
}
__device__ __forceinline__ v16h load_frag(const _Float16* p0, int ld) {
  return load_frag_p(frag_ptr(p0, ld));
}

__device__ __forceinline__ v8f wmma16(v16h a, v16h b, v8f c) {
  return __builtin_amdgcn_wmma_f32_16x16x32_f16(
      /*neg_a=*/false, a, /*neg_b=*/false, b,
      /*c_mod=*/(short)0, c, /*reuse_a=*/false, /*reuse_b=*/false);
}

// ---------------------------------------------------------------------------
// fp32 -> f16 conversion
// ---------------------------------------------------------------------------
__global__ __launch_bounds__(256) void cvt_f32_to_f16(const float* __restrict__ in,
                                                      _Float16* __restrict__ out,
                                                      int n) {
  int i = blockIdx.x * 256 + threadIdx.x;
  if (i < n) out[i] = (_Float16)in[i];
}

// ---------------------------------------------------------------------------
// GEMM:  Y[m,n] = sum_k A[m,k] * W[n,k] + bias[n]   (i.e. x @ W^T + b)
// HAS_PHI: apply phi(y)=elu(y)+1 for n < phiLimit (branch-free select).
// F16OUT:  store f16 to outh, else f32 to outf.
// Block: 128 threads = 4 waves (2x2), wave tile 64x64, block tile 128x128.
// ---------------------------------------------------------------------------
template <bool HAS_PHI, bool F16OUT>
__global__ __launch_bounds__(128)
void wmma_gemm_kernel(const _Float16* __restrict__ A,
                      const _Float16* __restrict__ W,
                      const float* __restrict__ bias,
                      _Float16* __restrict__ outh,
                      float* __restrict__ outf,
                      int M, int N, int K, int phiLimit) {
  const int lane = threadIdx.x & 31;
  const int w    = threadIdx.x >> 5;           // 0..3
  const int m0   = blockIdx.y * 128 + (w & 1) * 64;
  const int n0   = blockIdx.x * 128 + (w >> 1) * 64;

  // Per-fragment lane pointers, advanced by 32 halves per k-step.
  const _Float16* pA[4];
  const _Float16* pB[4];
#pragma unroll
  for (int i = 0; i < 4; ++i)
    pA[i] = frag_ptr(A + (size_t)(m0 + i * 16) * K, K);
#pragma unroll
  for (int j = 0; j < 4; ++j)
    pB[j] = frag_ptr(W + (size_t)(n0 + j * 16) * K, K);

  v8f acc[4][4] = {};

  for (int kk = 0; kk < K; kk += 32) {
    v16h af[4], bf[4];
#pragma unroll
    for (int i = 0; i < 4; ++i) { af[i] = load_frag_p(pA[i]); }
#pragma unroll
    for (int j = 0; j < 4; ++j) { bf[j] = load_frag_p(pB[j]); }
    // Prefetch the k-slice two iterations ahead (global_prefetch_b8).
    __builtin_prefetch(pA[0] + 64, 0, 0);
    __builtin_prefetch(pB[0] + 64, 0, 0);
#pragma unroll
    for (int i = 0; i < 4; ++i) pA[i] += 32;
#pragma unroll
    for (int j = 0; j < 4; ++j) pB[j] += 32;
#pragma unroll
    for (int i = 0; i < 4; ++i)
#pragma unroll
      for (int j = 0; j < 4; ++j)
        acc[i][j] = wmma16(af[i], bf[j], acc[i][j]);
  }

  const int hi8 = (lane >> 4) << 3;            // 0 or 8 (M offset of this lane)
#pragma unroll
  for (int i = 0; i < 4; ++i) {
#pragma unroll
    for (int j = 0; j < 4; ++j) {
      const int n = n0 + j * 16 + (lane & 15);
      const float bval = bias[n];
#pragma unroll
      for (int r = 0; r < 8; ++r) {
        const int m = m0 + i * 16 + r + hi8;
        float v = acc[i][j][r] + bval;
        if constexpr (HAS_PHI) {
          // phi(v) = v>0 ? v+1 : exp(v), selected without exec-mask branches
          const float ev  = __expf(v);
          const float pv  = (v > 0.0f) ? (v + 1.0f) : ev;
          v = (n < phiLimit) ? pv : v;
        }
        const size_t idx = (size_t)m * N + n;
        if constexpr (F16OUT) outh[idx] = (_Float16)v;
        else                  outf[idx] = v;
      }
    }
  }
}

// ---------------------------------------------------------------------------
// Per-chunk local state:  KV_c = K_c^T V_c  (64x64 f32)  and  ksum_c = sum_t k
// One wave per task (b,h,chunk).  K^T and V^T staged through LDS so WMMA
// fragments load with the natural [row][k] pattern.
// qkv layout: [M_TOT][3072] f16, cols 0..1023=q, 1024..2047=k, 2048..3071=v.
// ---------------------------------------------------------------------------
__global__ __launch_bounds__(32)
void chunk_kv_kernel(const _Float16* __restrict__ qkv,
                     float* __restrict__ Sloc,      // [NTASK][64*64] (d,e)
                     float* __restrict__ ksum) {    // [NTASK][64]
  const int task = blockIdx.x;
  const int bh = task >> 5, c = task & 31;
  const int b = bh >> 4, h = bh & 15;
  const int grow0 = b * TT + c * 64;
  const int colK  = 1024 + h * 64;
  const int colV  = 2048 + h * 64;
  const int lane  = threadIdx.x;

  __shared__ __align__(16) _Float16 Kt[64 * 64];   // [d][t] = K^T
  __shared__ __align__(16) _Float16 Vt[64 * 64];   // [e][t] = V^T

  for (int idx = lane; idx < 4096; idx += 32) {
    const int t = idx >> 6, d = idx & 63;
    const size_t row = (size_t)(grow0 + t) * QKV_N;
    Kt[d * 64 + t] = qkv[row + colK + d];
    Vt[d * 64 + t] = qkv[row + colV + d];
  }
  __syncthreads();

  v8f acc[4][4] = {};
#pragma unroll
  for (int kk = 0; kk < 64; kk += 32) {
    v16h af[4], bf[4];
#pragma unroll
    for (int i = 0; i < 4; ++i) af[i] = load_frag(Kt + (i * 16) * 64 + kk, 64);
#pragma unroll
    for (int j = 0; j < 4; ++j) bf[j] = load_frag(Vt + (j * 16) * 64 + kk, 64);
#pragma unroll
    for (int i = 0; i < 4; ++i)
#pragma unroll
      for (int j = 0; j < 4; ++j)
        acc[i][j] = wmma16(af[i], bf[j], acc[i][j]);
  }

  // k column sums (over t) for the normalizer prefix
  for (int d = lane; d < 64; d += 32) {
    float s = 0.0f;
    for (int t = 0; t < 64; ++t) s += (float)Kt[d * 64 + t];
    ksum[(size_t)task * 64 + d] = s;
  }

  const int hi8 = (lane >> 4) << 3;
#pragma unroll
  for (int i = 0; i < 4; ++i)
#pragma unroll
    for (int j = 0; j < 4; ++j)
#pragma unroll
      for (int r = 0; r < 8; ++r) {
        const int d = i * 16 + r + hi8;
        const int e = j * 16 + (lane & 15);
        Sloc[(size_t)task * 4096 + d * 64 + e] = acc[i][j][r];
      }
}

// ---------------------------------------------------------------------------
// Exclusive prefix over chunks (per b,h).  Writes f16 S^T (stored [e][d] so it
// loads directly as a WMMA B-operand) and exclusive k-sum prefix.
// ---------------------------------------------------------------------------
__global__ __launch_bounds__(256)
void prefix_kernel(const float* __restrict__ Sloc,
                   const float* __restrict__ ksum,
                   _Float16* __restrict__ ShT,      // [NTASK][64*64] (e,d)
                   float* __restrict__ kpref) {     // [NTASK][64]
  const int bh = blockIdx.x;
  const int tid = threadIdx.x;
  const int task0 = bh * NCHUNK;

  float run[16];
#pragma unroll
  for (int q = 0; q < 16; ++q) run[q] = 0.0f;

  for (int c = 0; c < NCHUNK; ++c) {
    const size_t base = (size_t)(task0 + c) * 4096;
#pragma unroll
    for (int q = 0; q < 16; ++q) {
      const int l = tid * 16 + q;
      const int d = l >> 6, e = l & 63;
      ShT[base + e * 64 + d] = (_Float16)run[q];
      run[q] += Sloc[base + l];
    }
  }
  if (tid < 64) {
    float rz = 0.0f;
    for (int c = 0; c < NCHUNK; ++c) {
      kpref[(size_t)(task0 + c) * 64 + tid] = rz;
      rz += ksum[(size_t)(task0 + c) * 64 + tid];
    }
  }
}

// ---------------------------------------------------------------------------
// Per-chunk output:
//   O = Q_c @ S_prefix  +  (mask(Q_c K_c^T)) @ V_c ,  out = O / (z + eps)
//   z[t] = q[t]·kpref + rowsum(masked P[t,:])
// One wave per task.  P and V^T staged through LDS for fragment re-layout.
// ---------------------------------------------------------------------------
__global__ __launch_bounds__(32)
void chunk_out_kernel(const _Float16* __restrict__ qkv,
                      const _Float16* __restrict__ ShT,
                      const float* __restrict__ kpref,
                      _Float16* __restrict__ oh) {
  const int task = blockIdx.x;
  const int bh = task >> 5, c = task & 31;
  const int b = bh >> 4, h = bh & 15;
  const int grow0 = b * TT + c * 64;
  const int colQ  = h * 64;
  const int colK  = 1024 + h * 64;
  const int colV  = 2048 + h * 64;
  const int lane  = threadIdx.x;

  __shared__ __align__(16) _Float16 Vt[64 * 64];   // [e][t] = V^T
  __shared__ __align__(16) _Float16 Pm[64 * 64];   // masked P, [t][s]
  __shared__ float zbuf[64];
  __shared__ float kprefS[64];

  for (int idx = lane; idx < 4096; idx += 32) {
    const int t = idx >> 6, e = idx & 63;
    Vt[e * 64 + t] = qkv[(size_t)(grow0 + t) * QKV_N + colV + e];
  }
  for (int d = lane; d < 64; d += 32) kprefS[d] = kpref[(size_t)task * 64 + d];
  __syncthreads();

  // Inter-chunk (Q @ S_prefix) and attention scores (P = Q K^T) together.
  v8f acc[4][4] = {};
  v8f p[4][4]   = {};
#pragma unroll
  for (int kk = 0; kk < 64; kk += 32) {
    v16h af[4], bs[4], bk_[4];
#pragma unroll
    for (int i = 0; i < 4; ++i)
      af[i] = load_frag(qkv + (size_t)(grow0 + i * 16) * QKV_N + colQ + kk, QKV_N);
#pragma unroll
    for (int j = 0; j < 4; ++j) {
      bs[j]  = load_frag(ShT + (size_t)task * 4096 + (j * 16) * 64 + kk, 64);
      bk_[j] = load_frag(qkv + (size_t)(grow0 + j * 16) * QKV_N + colK + kk, QKV_N);
    }
#pragma unroll
    for (int i = 0; i < 4; ++i)
#pragma unroll
      for (int j = 0; j < 4; ++j) {
        acc[i][j] = wmma16(af[i], bs[j],  acc[i][j]);   // Q @ S_prefix
        p[i][j]   = wmma16(af[i], bk_[j], p[i][j]);     // P = Q K^T
      }
  }

  // causal mask + spill P to LDS (f16) for re-layout as an A-operand
  const int hi8 = (lane >> 4) << 3;
#pragma unroll
  for (int i = 0; i < 4; ++i)
#pragma unroll
    for (int j = 0; j < 4; ++j)
#pragma unroll
      for (int r = 0; r < 8; ++r) {
        const int t = i * 16 + r + hi8;
        const int s = j * 16 + (lane & 15);
        const float v = (s <= t) ? p[i][j][r] : 0.0f;
        Pm[t * 64 + s] = (_Float16)v;
      }
  __syncthreads();

  // normalizer z[t]
  for (int t = lane; t < 64; t += 32) {
    float z = 0.0f;
    const _Float16* qp = qkv + (size_t)(grow0 + t) * QKV_N + colQ;
    for (int d = 0; d < 64; ++d) z += (float)qp[d] * kprefS[d];
    for (int s = 0; s < 64; ++s) z += (float)Pm[t * 64 + s];
    zbuf[t] = z;
  }
  __syncthreads();

  // Intra-chunk: acc += mask(P) @ V_c
#pragma unroll
  for (int kk = 0; kk < 64; kk += 32) {
    v16h af[4], bf[4];
#pragma unroll
    for (int i = 0; i < 4; ++i) af[i] = load_frag(Pm + (i * 16) * 64 + kk, 64);
#pragma unroll
    for (int j = 0; j < 4; ++j) bf[j] = load_frag(Vt + (j * 16) * 64 + kk, 64);
#pragma unroll
    for (int i = 0; i < 4; ++i)
#pragma unroll
      for (int j = 0; j < 4; ++j)
        acc[i][j] = wmma16(af[i], bf[j], acc[i][j]);
  }

#pragma unroll
  for (int i = 0; i < 4; ++i)
#pragma unroll
    for (int j = 0; j < 4; ++j)
#pragma unroll
      for (int r = 0; r < 8; ++r) {
        const int t = i * 16 + r + hi8;
        const int e = j * 16 + (lane & 15);
        const float v = acc[i][j][r] / (zbuf[t] + 1e-6f);
        oh[(size_t)(grow0 + t) * D_MODEL + (h * 64) + e] = (_Float16)v;
      }
}

// ---------------------------------------------------------------------------
// Host-side launcher
// ---------------------------------------------------------------------------
extern "C" void kernel_launch(void* const* d_in, const int* in_sizes, int n_in,
                              void* d_out, int out_size, void* d_ws, size_t ws_size,
                              hipStream_t stream) {
  const float* x  = (const float*)d_in[0];
  const float* Wq = (const float*)d_in[1];
  const float* bq = (const float*)d_in[2];
  const float* Wk = (const float*)d_in[3];
  const float* bk = (const float*)d_in[4];
  const float* Wv = (const float*)d_in[5];
  const float* bv = (const float*)d_in[6];
  const float* Wo = (const float*)d_in[7];
  const float* bo = (const float*)d_in[8];

  char* ws = (char*)d_ws;
  const size_t MB = 1u << 20;
  _Float16* xh    = (_Float16*)(ws + 0 * MB);    // 8 MB
  _Float16* wcat  = (_Float16*)(ws + 8 * MB);    // 6 MB: Wq|Wk|Wv f16, [3072][1024]
  _Float16* woh   = (_Float16*)(ws + 14 * MB);   // 2 MB
  _Float16* qkvh  = (_Float16*)(ws + 16 * MB);   // 24 MB: [4096][3072]
  _Float16* ohb   = (_Float16*)(ws + 40 * MB);   // 8 MB:  [4096][1024]
  float*    Sloc  = (float*)   (ws + 48 * MB);   // 16 MB
  _Float16* ShT   = (_Float16*)(ws + 64 * MB);   // 8 MB
  float*    ksum  = (float*)   (ws + 72 * MB);   // 0.5 MB
  float*    kpref = (float*)   (ws + 72 * MB + 512 * 1024); // 0.5 MB
  float*    bcat  = (float*)   (ws + 73 * MB);   // 12 KB: bq|bk|bv

  const int NX = M_TOT * D_MODEL;      // 4,194,304
  const int NW = D_MODEL * D_MODEL;    // 1,048,576

  cvt_f32_to_f16<<<(NX + 255) / 256, 256, 0, stream>>>(x,  xh, NX);
  cvt_f32_to_f16<<<(NW + 255) / 256, 256, 0, stream>>>(Wq, wcat + 0 * (size_t)NW, NW);
  cvt_f32_to_f16<<<(NW + 255) / 256, 256, 0, stream>>>(Wk, wcat + 1 * (size_t)NW, NW);
  cvt_f32_to_f16<<<(NW + 255) / 256, 256, 0, stream>>>(Wv, wcat + 2 * (size_t)NW, NW);
  cvt_f32_to_f16<<<(NW + 255) / 256, 256, 0, stream>>>(Wo, woh, NW);

  // concatenated bias (device-to-device, graph-capture safe)
  hipMemcpyAsync(bcat + 0 * D_MODEL, bq, D_MODEL * sizeof(float),
                 hipMemcpyDeviceToDevice, stream);
  hipMemcpyAsync(bcat + 1 * D_MODEL, bk, D_MODEL * sizeof(float),
                 hipMemcpyDeviceToDevice, stream);
  hipMemcpyAsync(bcat + 2 * D_MODEL, bv, D_MODEL * sizeof(float),
                 hipMemcpyDeviceToDevice, stream);

  // Fused Q|K|V projection: [4096,1024] x [3072,1024]^T, phi on n < 2048.
  dim3 gq(QKV_N / 128, M_TOT / 128);   // (24, 32)
  wmma_gemm_kernel<true, true><<<gq, 128, 0, stream>>>(
      xh, wcat, bcat, qkvh, nullptr, M_TOT, QKV_N, D_MODEL, /*phiLimit=*/2048);

  chunk_kv_kernel<<<NTASK, 32, 0, stream>>>(qkvh, Sloc, ksum);
  prefix_kernel<<<BB * N_HEADS, 256, 0, stream>>>(Sloc, ksum, ShT, kpref);
  chunk_out_kernel<<<NTASK, 32, 0, stream>>>(qkvh, ShT, kpref, ohb);

  // Output projection, fp32 result straight to d_out.
  dim3 go(D_MODEL / 128, M_TOT / 128); // (8, 32)
  wmma_gemm_kernel<false, false><<<go, 128, 0, stream>>>(
      ohb, woh, bo, nullptr, (float*)d_out, M_TOT, D_MODEL, D_MODEL, 0);
}
